// CrossAttnBlock_84877143703797
// MI455X (gfx1250) — matrix operand order repaired and checked by reading, not compiled, not run on hardware
//
#include <hip/hip_runtime.h>
#include <hip/hip_bf16.h>

// ---------------------------------------------------------------------------
// CrossAttnBlock for MI455X (gfx1250): all GEMMs + attention via
// v_wmma_f32_16x16x32_f16 (wave32 WMMA), LDS-staged tiles (A-tile via
// GLOBAL_LOAD_ASYNC_TO_LDS_B128 + ASYNCcnt), flash softmax with DPP
// cross-lane reductions.
// ---------------------------------------------------------------------------

typedef __attribute__((ext_vector_type(16))) _Float16 v16h;
typedef __attribute__((ext_vector_type(8)))  float    v8f;
typedef __attribute__((ext_vector_type(4)))  float    f32x4;

#define WMMA_F16(a, b, c) \
  __builtin_amdgcn_wmma_f32_16x16x32_f16(false, (a), false, (b), (short)0, (c), false, false)

__device__ __forceinline__ v16h load_frag16(const _Float16* p0, const _Float16* p1) {
  union { v16h h; f32x4 q[2]; } u;
  u.q[0] = *(const f32x4*)p0;
  u.q[1] = *(const f32x4*)p1;
  return u.h;
}

// ---- CDNA5 async global->LDS copy (tracked with ASYNCcnt) -----------------
// LDS byte offset = low 32 bits of generic pointer.
__device__ __forceinline__ void async_load_b128(unsigned lds_off, const void* gptr) {
  asm volatile("global_load_async_to_lds_b128 %0, %1, off"
               :: "v"(lds_off), "v"(gptr)
               : "memory");
}
__device__ __forceinline__ void wait_async0() {
  asm volatile("s_wait_asynccnt 0" ::: "memory");
}

// ---- DPP helpers: butterfly reduction across a 16-lane row ----------------
constexpr int dpp8_xor_sel(int m) {
  int s = 0;
  for (int i = 0; i < 8; ++i) s |= ((i ^ m) & 7) << (3 * i);
  return s;
}
enum : int {
  DPP8_XOR1 = dpp8_xor_sel(1),
  DPP8_XOR2 = dpp8_xor_sel(2),
  DPP8_XOR4 = dpp8_xor_sel(4),
  DPP16_ROW_ROR8 = 0x128            // row_ror:8 == lane ^ 8 within 16-lane row
};

__device__ __forceinline__ float red_max16(float x) {
  union { float f; int i; } u, v;
  u.f = x;
  v.i = __builtin_amdgcn_mov_dpp8(u.i, DPP8_XOR1); u.f = fmaxf(u.f, v.f);
  v.i = __builtin_amdgcn_mov_dpp8(u.i, DPP8_XOR2); u.f = fmaxf(u.f, v.f);
  v.i = __builtin_amdgcn_mov_dpp8(u.i, DPP8_XOR4); u.f = fmaxf(u.f, v.f);
  v.i = __builtin_amdgcn_update_dpp(0, u.i, DPP16_ROW_ROR8, 0xf, 0xf, true);
  u.f = fmaxf(u.f, v.f);
  return u.f;
}

__device__ __forceinline__ float red_sum16(float x) {
  union { float f; int i; } u, v;
  u.f = x;
  v.i = __builtin_amdgcn_mov_dpp8(u.i, DPP8_XOR1); u.f += v.f;
  v.i = __builtin_amdgcn_mov_dpp8(u.i, DPP8_XOR2); u.f += v.f;
  v.i = __builtin_amdgcn_mov_dpp8(u.i, DPP8_XOR4); u.f += v.f;
  v.i = __builtin_amdgcn_update_dpp(0, u.i, DPP16_ROW_ROR8, 0xf, 0xf, true);
  u.f += v.f;
  return u.f;
}

// ---------------------------------------------------------------------------
// Problem constants (match reference setup_inputs)
// ---------------------------------------------------------------------------
#define CB   8
#define CLQ  2048
#define CLKV 512
#define CH   1024
#define CNH  16
#define CDH  64
#define CH2  2048   // 2*H

// ---------------------------------------------------------------------------
// Kernel 1: ss = silu(t) @ Wss + bss    (B x 2H)
// ---------------------------------------------------------------------------
__global__ __launch_bounds__(256) void ss_kernel(
    const float* __restrict__ t, const float* __restrict__ W,
    const float* __restrict__ bvec, float* __restrict__ ss) {
  __shared__ float st[CH];
  int tid = threadIdx.x;
  int b = blockIdx.y;
  int j = blockIdx.x * 256 + tid;
  for (int i = tid; i < CH; i += 256) {
    float c = t[(size_t)b * CH + i];
    st[i] = c / (1.0f + __expf(-c));      // silu
  }
  __syncthreads();
  float acc = bvec[j];
  for (int k = 0; k < CH; ++k)
    acc = fmaf(st[k], W[(size_t)k * CH2 + j], acc);
  ss[(size_t)b * CH2 + j] = acc;
}

// ---------------------------------------------------------------------------
// Kernel 2: fp32 -> f16 cast (weights)
// ---------------------------------------------------------------------------
__global__ __launch_bounds__(256) void cast_h(
    const float* __restrict__ x, _Float16* __restrict__ y, int n) {
  int i = blockIdx.x * 256 + threadIdx.x;
  if (i < n) y[i] = (_Float16)x[i];
}

// ---------------------------------------------------------------------------
// Kernel 3: AdaLN: y = (1+scale)*normalize(x) + bias, output f16.
// One 256-thread block per row of H=1024.
// ---------------------------------------------------------------------------
__global__ __launch_bounds__(256) void ada_ln_kernel(
    const float* __restrict__ x, const float* __restrict__ ss,
    _Float16* __restrict__ y, int L) {
  int row = blockIdx.x;              // 0 .. B*L-1
  int b = row / L;
  int tid = threadIdx.x;
  const float* xr = x + (size_t)row * CH;

  float s1 = 0.f, s2 = 0.f;
#pragma unroll
  for (int j = tid; j < CH; j += 256) {
    float v = xr[j];
    s1 += v;
    s2 += v * v;
  }
#pragma unroll
  for (int off = 16; off > 0; off >>= 1) {
    s1 += __shfl_xor(s1, off, 32);
    s2 += __shfl_xor(s2, off, 32);
  }
  __shared__ float r1[8], r2[8];
  int wv = tid >> 5, ln = tid & 31;
  if (ln == 0) { r1[wv] = s1; r2[wv] = s2; }
  __syncthreads();
  if (wv == 0) {
    float a = (ln < 8) ? r1[ln] : 0.f;
    float c = (ln < 8) ? r2[ln] : 0.f;
#pragma unroll
    for (int off = 4; off > 0; off >>= 1) {
      a += __shfl_xor(a, off, 32);
      c += __shfl_xor(c, off, 32);
    }
    if (ln == 0) { r1[0] = a; r2[0] = c; }
  }
  __syncthreads();
  float mu   = r1[0] * (1.f / CH);
  float var  = r2[0] * (1.f / CH) - mu * mu;
  float rinv = rsqrtf(var + 1e-5f);

  const float* ssb = ss + (size_t)b * CH2;
  _Float16* yr = y + (size_t)row * CH;
#pragma unroll
  for (int j = tid; j < CH; j += 256) {
    float hn = (xr[j] - mu) * rinv;
    float sc = ssb[j];
    float bi = ssb[CH + j];
    yr[j] = (_Float16)((1.0f + sc) * hn + bi);
  }
}

// ---------------------------------------------------------------------------
// Kernel 4: tiled f16 WMMA GEMM: C = A(MxK) * W(KxN) + bias
//   block tile 128x128, BK=32, 8 waves -> 2x4 wave grid, 4x2 WMMA tiles/wave.
//   A-tile staged with GLOBAL_LOAD_ASYNC_TO_LDS_B128 (ASYNCcnt).
//   RESID=false: out f16.   RESID=true: out f32 with residual add.
// ---------------------------------------------------------------------------
#define BM 128
#define BN 128
#define BK 32

template <bool RESID>
__global__ __launch_bounds__(256) void gemm_f16(
    const _Float16* __restrict__ A, const _Float16* __restrict__ W,
    const float* __restrict__ bias, const float* __restrict__ resid,
    _Float16* __restrict__ outh, float* __restrict__ outf,
    int M, int N, int K) {
  __shared__ __align__(16) _Float16 sA[BM * BK];       // [m][k]
  __shared__ __align__(16) _Float16 sB[BN * BK];       // transposed: [n][k]

  int tid  = threadIdx.x;
  int wave = tid >> 5, lane = tid & 31;
  int half_ = lane >> 4, l16 = lane & 15;
  int waveM = wave >> 2, waveN = wave & 3;             // 2 x 4 wave grid
  int blockM = blockIdx.y * BM, blockN = blockIdx.x * BN;

  // Per-thread A-tile chunk coordinates (2 x 16B chunks per thread)
  int arow0 = (tid * 2) >> 3;           // chunk c=2*tid   -> row c/4? (see below)
  // chunk id c in 0..511: row = c>>2, kk = (c&3)*8
  unsigned a_lds[2];
  size_t   a_goff[2];
#pragma unroll
  for (int i = 0; i < 2; ++i) {
    int c   = tid + i * 256;
    int row = c >> 2;
    int kk  = (c & 3) * 8;
    a_lds[i]  = (unsigned)(size_t)&sA[row * BK + kk];
    a_goff[i] = (size_t)(blockM + row) * K + kk;
  }
  (void)arow0;

  v8f acc[4][2] = {};

  for (int k0 = 0; k0 < K; k0 += BK) {
    // Stage A asynchronously: global -> LDS without VGPR round trip
#pragma unroll
    for (int i = 0; i < 2; ++i)
      async_load_b128(a_lds[i], &A[a_goff[i] + k0]);

    // Stage B transposed: sB[n][k] = W[k0+k][blockN+n] (coalesced along n)
#pragma unroll
    for (int i = 0; i < 16; ++i) {
      int e  = tid + i * 256;                           // 0..4095
      int kk = e >> 7;
      int nn = e & 127;
      sB[nn * BK + kk] = W[(size_t)(k0 + kk) * N + blockN + nn];
    }
    if (k0 + BK < K) {
      // L2 prefetch of next weight block (global_prefetch_b8)
      __builtin_prefetch(&W[(size_t)(k0 + BK) * N + blockN + (tid & 127)], 0, 1);
    }
    wait_async0();                     // A tile landed in LDS
    __syncthreads();

    // B fragments: lane=col n, K = {half*8+0..7, 16+half*8+0..7}
    v16h bfr[2];
#pragma unroll
    for (int sn = 0; sn < 2; ++sn) {
      int n = waveN * 32 + sn * 16 + l16;
      bfr[sn] = load_frag16(&sB[n * BK + half_ * 8],
                            &sB[n * BK + 16 + half_ * 8]);
    }
    // A fragments: lane=row m, K = {half*8+0..7, 16+half*8+0..7}
#pragma unroll
    for (int sm = 0; sm < 4; ++sm) {
      int m = waveM * 64 + sm * 16 + l16;
      v16h afr = load_frag16(&sA[m * BK + half_ * 8],
                             &sA[m * BK + 16 + half_ * 8]);
#pragma unroll
      for (int sn = 0; sn < 2; ++sn)
        acc[sm][sn] = WMMA_F16(afr, bfr[sn], acc[sm][sn]);
    }
    __syncthreads();
  }

  // Epilogue: C layout -> lane holds col l16, rows r + 8*half_
#pragma unroll
  for (int sm = 0; sm < 4; ++sm) {
#pragma unroll
    for (int sn = 0; sn < 2; ++sn) {
      int n = blockN + waveN * 32 + sn * 16 + l16;
      float bv = bias[n];
#pragma unroll
      for (int r = 0; r < 8; ++r) {
        int m = blockM + waveM * 64 + sm * 16 + r + 8 * half_;
        float val = acc[sm][sn][r] + bv;
        if (RESID)
          outf[(size_t)m * N + n] = resid[(size_t)m * N + n] + val;
        else
          outh[(size_t)m * N + n] = (_Float16)val;
      }
    }
  }
}

// ---------------------------------------------------------------------------
// Kernel 5: flash attention, per-wave 16 q-rows over Lkv=512, dh=64.
//   scores = (Q K^T)/8 via 2x WMMA per 16-col tile; online softmax (DPP
//   reductions); P re-laid out through LDS; PV via WMMA (K=32 kv per step).
// ---------------------------------------------------------------------------
__global__ __launch_bounds__(128) void attn_kernel(
    const _Float16* __restrict__ q,   // (B, Lq, H) f16
    const _Float16* __restrict__ kv,  // (B, Lkv, 2H) f16: cols [0,H)=k, [H,2H)=v
    _Float16* __restrict__ out) {     // (B, Lq, H) f16
  __shared__ __align__(16) _Float16 sP[4][16 * 32];   // per-wave P tile [m][k]
  __shared__ __align__(16) _Float16 sV[4][64 * 32];   // per-wave V^T tile [n][k]

  int wave = threadIdx.x >> 5, lane = threadIdx.x & 31;
  int half_ = lane >> 4, l16 = lane & 15;
  int b = blockIdx.z, h = blockIdx.y;
  int qbase = blockIdx.x * 64 + wave * 16;

  // Q A-fragments for dh=64 -> two K=32 chunks; contiguous 16B pieces.
  v16h qfr[2];
  {
    const _Float16* qrow = q + ((size_t)(b * CLQ + qbase + l16) * CH + h * CDH);
#pragma unroll
    for (int c = 0; c < 2; ++c)
      qfr[c] = load_frag16(qrow + c * 32 + half_ * 8,
                           qrow + c * 32 + 16 + half_ * 8);
  }

  v8f O[4] = {};
  float mrow[8], lrow[8];
#pragma unroll
  for (int r = 0; r < 8; ++r) { mrow[r] = -3.0e38f; lrow[r] = 0.f; }

  for (int kt = 0; kt < CLKV; kt += 32) {
    // ---- scores: two 16-col tiles (kv rows kt..kt+15, kt+16..kt+31) ----
    v8f s0 = {}, s1 = {};
#pragma unroll
    for (int c = 0; c < 2; ++c) {
      const _Float16* kb0 =
          kv + ((size_t)(b * CLKV + kt + l16) * CH2 + h * CDH + c * 32);
      const _Float16* kb1 =
          kv + ((size_t)(b * CLKV + kt + 16 + l16) * CH2 + h * CDH + c * 32);
      v16h kf0 = load_frag16(kb0 + half_ * 8, kb0 + 16 + half_ * 8);
      v16h kf1 = load_frag16(kb1 + half_ * 8, kb1 + 16 + half_ * 8);
      s0 = WMMA_F16(qfr[c], kf0, s0);
      s1 = WMMA_F16(qfr[c], kf1, s1);
    }
    s0 = s0 * 0.125f;   // 1/sqrt(64)
    s1 = s1 * 0.125f;

    // ---- stage V tile transposed: sV[n][k]; b128 global reads ----
#pragma unroll
    for (int i = 0; i < 8; ++i) {
      int c  = lane + i * 32;     // chunk 0..255 (8 f16 each)
      int kk = c >> 3;            // kv row in tile (32 rows)
      int n0 = (c & 7) * 8;       // first dh col of chunk
      union { f32x4 v4; _Float16 h[8]; } u;
      u.v4 = *(const f32x4*)&kv[(size_t)(b * CLKV + kt + kk) * CH2 + CH +
                                h * CDH + n0];
#pragma unroll
      for (int j = 0; j < 8; ++j)
        sV[wave][(n0 + j) * 32 + kk] = u.h[j];
    }

    // ---- online softmax over the 32 new kv columns (DPP reductions) ----
#pragma unroll
    for (int r = 0; r < 8; ++r) {
      float a0 = s0[r], a1 = s1[r];
      float mx = red_max16(fmaxf(a0, a1));               // row max, 16 lanes
      float mnew  = fmaxf(mrow[r], mx);
      float alpha = __expf(mrow[r] - mnew);
      float p0 = __expf(a0 - mnew);
      float p1 = __expf(a1 - mnew);
      float ps = red_sum16(p0 + p1);                     // row sum, 16 lanes
      lrow[r] = lrow[r] * alpha + ps;
      mrow[r] = mnew;
#pragma unroll
      for (int sn = 0; sn < 4; ++sn) O[sn][r] *= alpha;
      int m = r + 8 * half_;
      sP[wave][m * 32 + l16]      = (_Float16)p0;        // C-layout -> [m][k]
      sP[wave][m * 32 + 16 + l16] = (_Float16)p1;
    }

    // ---- O += P(16x32) * V(32x64) ----
    v16h pfr = load_frag16(&sP[wave][l16 * 32 + half_ * 8],
                           &sP[wave][l16 * 32 + 16 + half_ * 8]);
#pragma unroll
    for (int sn = 0; sn < 4; ++sn) {
      int n = sn * 16 + l16;
      v16h vfr = load_frag16(&sV[wave][n * 32 + half_ * 8],
                             &sV[wave][n * 32 + 16 + half_ * 8]);
      O[sn] = WMMA_F16(pfr, vfr, O[sn]);
    }
  }

  // ---- normalize and write out (f16) ----
  float linv[8];
#pragma unroll
  for (int r = 0; r < 8; ++r) linv[r] = 1.0f / lrow[r];
#pragma unroll
  for (int sn = 0; sn < 4; ++sn) {
    int col = h * CDH + sn * 16 + l16;
#pragma unroll
    for (int r = 0; r < 8; ++r) {
      int row = qbase + r + 8 * half_;
      out[(size_t)(b * CLQ + row) * CH + col] = (_Float16)(O[sn][r] * linv[r]);
    }
  }
}

// ---------------------------------------------------------------------------
// Host-side orchestration
// ---------------------------------------------------------------------------
extern "C" void kernel_launch(void* const* d_in, const int* in_sizes, int n_in,
                              void* d_out, int out_size, void* d_ws, size_t ws_size,
                              hipStream_t stream) {
  (void)in_sizes; (void)n_in; (void)out_size; (void)ws_size;

  const float* x_q    = (const float*)d_in[0];
  const float* x_kv   = (const float*)d_in[1];
  const float* t_vec  = (const float*)d_in[2];
  const float* Wq     = (const float*)d_in[3];
  const float* bq     = (const float*)d_in[4];
  const float* Wkv    = (const float*)d_in[5];
  const float* bkv    = (const float*)d_in[6];
  const float* Wp     = (const float*)d_in[7];
  const float* bp     = (const float*)d_in[8];
  const float* Wss_q  = (const float*)d_in[9];
  const float* bss_q  = (const float*)d_in[10];
  const float* Wss_kv = (const float*)d_in[11];
  const float* bss_kv = (const float*)d_in[12];
  float* outp = (float*)d_out;

  char* ws = (char*)d_ws;
  size_t off = 0;
  auto bump = [&](size_t bytes) -> void* {
    void* p = ws + off;
    off = (off + bytes + 255) & ~(size_t)255;
    return p;
  };
  _Float16* Wq_h   = (_Float16*)bump((size_t)CH * CH * 2);
  _Float16* Wkv_h  = (_Float16*)bump((size_t)CH * CH2 * 2);
  _Float16* Wp_h   = (_Float16*)bump((size_t)CH * CH * 2);
  float*    ss_q   = (float*)bump((size_t)CB * CH2 * 4);
  float*    ss_kv  = (float*)bump((size_t)CB * CH2 * 4);
  _Float16* xqn    = (_Float16*)bump((size_t)CB * CLQ * CH * 2);
  _Float16* xkvn   = (_Float16*)bump((size_t)CB * CLKV * CH * 2);
  _Float16* q_h    = (_Float16*)bump((size_t)CB * CLQ * CH * 2);
  _Float16* kv_h   = (_Float16*)bump((size_t)CB * CLKV * CH2 * 2);
  _Float16* attn_h = (_Float16*)bump((size_t)CB * CLQ * CH * 2);

  // 1) ss = silu(t) @ Wss + bss
  ss_kernel<<<dim3(CH2 / 256, CB), 256, 0, stream>>>(t_vec, Wss_q,  bss_q,  ss_q);
  ss_kernel<<<dim3(CH2 / 256, CB), 256, 0, stream>>>(t_vec, Wss_kv, bss_kv, ss_kv);

  // 2) cast weights to f16
  cast_h<<<(CH * CH  + 255) / 256, 256, 0, stream>>>(Wq,  Wq_h,  CH * CH);
  cast_h<<<(CH * CH2 + 255) / 256, 256, 0, stream>>>(Wkv, Wkv_h, CH * CH2);
  cast_h<<<(CH * CH  + 255) / 256, 256, 0, stream>>>(Wp,  Wp_h,  CH * CH);

  // 3) AdaLN -> f16 activations
  ada_ln_kernel<<<CB * CLQ,  256, 0, stream>>>(x_q,  ss_q,  xqn,  CLQ);
  ada_ln_kernel<<<CB * CLKV, 256, 0, stream>>>(x_kv, ss_kv, xkvn, CLKV);

  // 4) Q = xqn @ Wq + bq    (M=16384, N=1024, K=1024)
  gemm_f16<false><<<dim3(CH / BN, (CB * CLQ) / BM), 256, 0, stream>>>(
      xqn, Wq_h, bq, nullptr, q_h, nullptr, CB * CLQ, CH, CH);
  // 5) KV = xkvn @ Wkv + bkv (M=4096, N=2048, K=1024); cols [0,H)=k, [H,2H)=v
  gemm_f16<false><<<dim3(CH2 / BN, (CB * CLKV) / BM), 256, 0, stream>>>(
      xkvn, Wkv_h, bkv, nullptr, kv_h, nullptr, CB * CLKV, CH2, CH);

  // 6) attention
  attn_kernel<<<dim3(CLQ / 64, CNH, CB), 128, 0, stream>>>(q_h, kv_h, attn_h);

  // 7) out = x_q + attn @ Wp + bp  (f32, with residual)
  gemm_f16<true><<<dim3(CH / BN, (CB * CLQ) / BM), 256, 0, stream>>>(
      attn_h, Wp_h, bp, x_q, nullptr, outp, CB * CLQ, CH, CH);
}
